// AttAggFME_893353197690
// MI455X (gfx1250) — matrix operand order repaired and checked by reading, not compiled, not run on hardware
//
#include <hip/hip_runtime.h>
#include <hip/hip_bf16.h>
#include <stdint.h>

// ---------------------------------------------------------------------------
// Types for CDNA5 WMMA (wave32)
// ---------------------------------------------------------------------------
typedef __bf16 bf16_t;
typedef __attribute__((ext_vector_type(16))) __bf16    v16bf;
typedef __attribute__((ext_vector_type(8)))  float     v8f;
typedef uint32_t u32x4 __attribute__((ext_vector_type(4)));

#define HW  4096   // 64*64
#define WD  64
#define NB  4
#define NP  (NB*HW)

__device__ __forceinline__ float lrelu_f(float v) { return v > 0.f ? v : 0.1f * v; }

union AFrag { v16bf v; uint32_t u[8]; };
union BFrag { v16bf v; u32x4 q[2]; uint32_t u[8]; };

// A-matrix 16x32 (MxK) bf16 fragment. `row` points at the lane's M-row (K
// contiguous in memory), hi = (lane>=16).
//   vg 0..7 : K = hi*16 + vg*2 + {0,1}
__device__ __forceinline__ v16bf loadA32(const bf16_t* __restrict__ row, int hi) {
  AFrag a;
  const uint32_t* p = (const uint32_t*)row;
  const int ho = hi << 3;  // hi*16 elements == hi*8 dwords
#pragma unroll
  for (int vg = 0; vg < 8; ++vg) {
    const int k2 = (vg << 1);
    a.u[vg] = p[(k2 >> 1) + ho];
  }
  return a.v;
}

// B-matrix 32x16 (KxN) bf16 fragment. Lane holds column N = lane&15;
// lanes 0-15 hold K=0..15, lanes 16-31 hold K=16..31 (contiguous in memory).
// `col` must already include the +16-element offset for lanes>=16, 32B aligned.
__device__ __forceinline__ v16bf loadB32(const bf16_t* __restrict__ col) {
  BFrag b;
  const u32x4* p = (const u32x4*)col;
  b.q[0] = p[0];
  b.q[1] = p[1];
  return b.v;
}

// ---------------------------------------------------------------------------
// Generic implicit-GEMM WMMA convolution (channels-last bf16 in/out).
// One wave -> (OCT*16) output channels x 16 pixels: the B (im2col) fragment is
// loaded once per K-step and reused across OCT WMMAs. Out-of-bounds taps read
// a zeroed page via per-lane address select (no EXEC divergence in the loop).
//   X   : [NP][icPad]  bf16, channels-last, zero-padded channels
//   Zpg : >= (icPad+16) bf16 zeros
//   Wp  : [OCpad][KH][KW][icPad] bf16
//   Y   : [NP][ocStride] bf16 at channel offset ocOffset (optional)
//   Yf  : same layout f32 (optional)
// ---------------------------------------------------------------------------
template <int OCT>
__global__ void __launch_bounds__(32)
wmma_conv_kernel(const bf16_t* __restrict__ X, int icPad,
                 const bf16_t* __restrict__ Zpg,
                 const bf16_t* __restrict__ Wp,
                 const float* __restrict__ bias,
                 bf16_t* __restrict__ Y, float* __restrict__ Yf,
                 int ocStride, int ocOffset, int OC,
                 int KH, int KW, int dil, int relu)
{
  const int lane = threadIdx.x;
  const int lo16 = lane & 15;
  const int hi   = (lane >> 4) & 1;
  const int pixBase = blockIdx.x << 4;
  const int ocBase  = blockIdx.y * (OCT * 16);
  const int b  = pixBase >> 12;
  const int sp = pixBase & (HW - 1);
  const int y0 = sp >> 6;
  const int x0 = sp & (WD - 1);
  const int px = x0 + lo16;              // all 16 pixels share row y0 (64%16==0)
  const int pad = ((KH - 1) * dil) >> 1;

  v8f acc[OCT];
#pragma unroll
  for (int t = 0; t < OCT; ++t) acc[t] = (v8f){0.f,0.f,0.f,0.f,0.f,0.f,0.f,0.f};

  // A-fragment row pointers (weights), one per oc tile
  const bf16_t* wrow0 = Wp + (size_t)(ocBase + lo16) * KH * KW * icPad;

  for (int kh = 0; kh < KH; ++kh) {
    const int ys = y0 + kh * dil - pad;
    const bool yok = (ys >= 0) && (ys < WD);
    for (int kw = 0; kw < KW; ++kw) {
      const int xs = px + kw * dil - pad;
      const bool ok = yok && (xs >= 0) && (xs < WD);
      // per-lane address select: OOB lanes read the zero page
      const bf16_t* xrow =
          (ok ? (X + (size_t)(b * HW + ys * WD + xs) * icPad) : Zpg) + (hi << 4);
      const size_t woff = (size_t)(kh * KW + kw) * icPad;
      for (int ic0 = 0; ic0 < icPad; ic0 += 32) {
        v16bf bb = loadB32(xrow + ic0);
#pragma unroll
        for (int t = 0; t < OCT; ++t) {
          v16bf a = loadA32(wrow0 + (size_t)t * 16 * KH * KW * icPad + woff + ic0, hi);
          acc[t] = __builtin_amdgcn_wmma_f32_16x16x32_bf16(false, a, false, bb,
                                                           (short)0, acc[t], false, false);
        }
      }
    }
  }

  const int pix = pixBase + lo16;        // C-fragment column N = pixel
#pragma unroll
  for (int t = 0; t < OCT; ++t) {
#pragma unroll
    for (int r = 0; r < 8; ++r) {
      const int M  = r + (hi << 3);
      const int oc = ocBase + t * 16 + M;
      if (oc < OC) {
        float v = acc[t][r] + (bias ? bias[oc] : 0.f);
        if (relu) v = lrelu_f(v);
        if (Y)  Y [(size_t)pix * ocStride + ocOffset + oc] = (bf16_t)v;
        if (Yf) Yf[(size_t)pix * ocStride + ocOffset + oc] = v;
      }
    }
  }
}

// ---------------------------------------------------------------------------
// Flash attention, one wave = 16 query rows, KV tiles of 32, d = 128.
//  QK : [NP][256] bf16   (q = ch 0..127, k = ch 128..255)
//  VT : [NB][128][HW] bf16   (V transposed: dim-major)
//  AGG: [NP][128] f32    out = softmax(scale * QK^T) * V
// ---------------------------------------------------------------------------
__global__ void __launch_bounds__(32)
flash_attn_kernel(const bf16_t* __restrict__ QK,
                  const bf16_t* __restrict__ VT,
                  float* __restrict__ AGG)
{
  __shared__ bf16_t Pl[16 * 32];         // P tile staged C-layout -> A-layout
  const int lane = threadIdx.x;
  const int lo16 = lane & 15;
  const int hi   = (lane >> 4) & 1;
  const int b     = blockIdx.y;
  const int qBase = blockIdx.x << 4;
  const float scale = 0.08838834764831845f;   // 128^-0.5

  // preload Q as 4 A-fragments (K = 128 = 4*32)
  v16bf Aq[4];
  {
    const bf16_t* qrow = QK + ((size_t)(b * HW) + qBase + lo16) * 256;
#pragma unroll
    for (int kc = 0; kc < 4; ++kc) Aq[kc] = loadA32(qrow + kc * 32, hi);
  }

  v8f O[8];
#pragma unroll
  for (int n = 0; n < 8; ++n) O[n] = (v8f){0.f,0.f,0.f,0.f,0.f,0.f,0.f,0.f};
  float mrow[8], lrow[8];
#pragma unroll
  for (int r = 0; r < 8; ++r) { mrow[r] = -1e30f; lrow[r] = 0.f; }

  for (int kv = 0; kv < HW; kv += 32) {
    v8f S0 = {0.f,0.f,0.f,0.f,0.f,0.f,0.f,0.f};
    v8f S1 = {0.f,0.f,0.f,0.f,0.f,0.f,0.f,0.f};
    const bf16_t* krow0 = QK + ((size_t)(b * HW) + kv + lo16) * 256 + 128 + (hi << 4);
    const bf16_t* krow1 = krow0 + (size_t)16 * 256;
#pragma unroll
    for (int kc = 0; kc < 4; ++kc) {
      v16bf B0 = loadB32(krow0 + kc * 32);
      S0 = __builtin_amdgcn_wmma_f32_16x16x32_bf16(false, Aq[kc], false, B0,
                                                   (short)0, S0, false, false);
      v16bf B1 = loadB32(krow1 + kc * 32);
      S1 = __builtin_amdgcn_wmma_f32_16x16x32_bf16(false, Aq[kc], false, B1,
                                                   (short)0, S1, false, false);
    }

    // online softmax per row (row stats live per 16-lane half-group)
#pragma unroll
    for (int r = 0; r < 8; ++r) {
      const float s0 = S0[r] * scale;
      const float s1 = S1[r] * scale;
      float mx = fmaxf(s0, s1);
#pragma unroll
      for (int msk = 1; msk < 16; msk <<= 1) mx = fmaxf(mx, __shfl_xor(mx, msk, 32));
      const float mnew  = fmaxf(mrow[r], mx);
      const float alpha = __expf(mrow[r] - mnew);
      const float p0 = __expf(s0 - mnew);
      const float p1 = __expf(s1 - mnew);
      float ps = p0 + p1;
#pragma unroll
      for (int msk = 1; msk < 16; msk <<= 1) ps += __shfl_xor(ps, msk, 32);
      lrow[r] = lrow[r] * alpha + ps;
      mrow[r] = mnew;
#pragma unroll
      for (int n = 0; n < 8; ++n) O[n][r] *= alpha;
      const int row = r + (hi << 3);
      Pl[row * 32 + lo16]      = (bf16_t)p0;
      Pl[row * 32 + 16 + lo16] = (bf16_t)p1;
    }
    __syncthreads();

    // reload P as an A-fragment (16 queries x 32 keys)
    v16bf Pa;
    {
      AFrag a;
      const uint32_t* pp = (const uint32_t*)(Pl + lo16 * 32);
#pragma unroll
      for (int vg = 0; vg < 8; ++vg) {
        const int k2 = (hi << 4) + (vg << 1);
        a.u[vg] = pp[k2 >> 1];
      }
      Pa = a.v;
    }
    __syncthreads();

    // O += P * V   (8 N-blocks of 16 dims; VT gives contiguous K loads)
    const bf16_t* vcol = VT + ((size_t)(b * 128) + lo16) * HW + kv + (hi << 4);
#pragma unroll
    for (int n = 0; n < 8; ++n) {
      v16bf Bv = loadB32(vcol + (size_t)n * 16 * HW);
      O[n] = __builtin_amdgcn_wmma_f32_16x16x32_bf16(false, Pa, false, Bv,
                                                     (short)0, O[n], false, false);
    }
  }

#pragma unroll
  for (int n = 0; n < 8; ++n) {
#pragma unroll
    for (int r = 0; r < 8; ++r) {
      const int M = r + (hi << 3);
      AGG[((size_t)(b * HW) + qBase + M) * 128 + n * 16 + lo16] = O[n][r] / lrow[r];
    }
  }
}

// ---------------------------------------------------------------------------
// Packing / layout / small kernels
// ---------------------------------------------------------------------------
__global__ void zero_fill_kernel(bf16_t* __restrict__ p, int n)
{
  int idx = blockIdx.x * blockDim.x + threadIdx.x;
  if (idx < n) p[idx] = (bf16_t)0.f;
}

__global__ void pack_w_kernel(const float* __restrict__ w, bf16_t* __restrict__ wp,
                              int OC, int OCpad, int IC, int ICpad, int KHW, int n)
{
  int idx = blockIdx.x * blockDim.x + threadIdx.x;
  if (idx >= n) return;
  const int ic = idx % ICpad;
  const int t  = idx / ICpad;
  const int k  = t % KHW;
  const int oc = t / KHW;
  float v = 0.f;
  if (oc < OC && ic < IC) v = w[((size_t)oc * IC + ic) * KHW + k];
  wp[idx] = (bf16_t)v;
}

// NCHW f32 -> channels-last bf16 with channel padding
__global__ void pack_x_kernel(const float* __restrict__ x, bf16_t* __restrict__ xp,
                              int fullC, int chOff, int C, int Cpad, int n)
{
  int idx = blockIdx.x * blockDim.x + threadIdx.x;
  if (idx >= n) return;
  const int c   = idx % Cpad;
  const int pix = idx / Cpad;
  const int b   = pix >> 12;
  const int sp  = pix & (HW - 1);
  float v = 0.f;
  if (c < C) v = x[((size_t)(b * fullC) + chOff + c) * HW + sp];
  xp[idx] = (bf16_t)v;
}

// convf1: 7x7, IC=2 -> OC=128, lrelu, NCHW f32 in -> channels-last bf16 out
__global__ void convf1_kernel(const float* __restrict__ flow, const float* __restrict__ w,
                              const float* __restrict__ bias, bf16_t* __restrict__ out)
{
  int idx = blockIdx.x * blockDim.x + threadIdx.x;
  if (idx >= NP * 128) return;
  const int oc  = idx & 127;
  const int pix = idx >> 7;
  const int b = pix >> 12, sp = pix & (HW - 1);
  const int y = sp >> 6, x = sp & 63;
  float acc = bias[oc];
  for (int ic = 0; ic < 2; ++ic)
    for (int ky = 0; ky < 7; ++ky) {
      const int ys = y + ky - 3; if (ys < 0 || ys >= WD) continue;
      for (int kx = 0; kx < 7; ++kx) {
        const int xs = x + kx - 3; if (xs < 0 || xs >= WD) continue;
        acc += flow[(size_t)(b * 2 + ic) * HW + ys * WD + xs] *
               w[((oc * 2 + ic) * 7 + ky) * 7 + kx];
      }
    }
  out[(size_t)pix * 128 + oc] = (bf16_t)lrelu_f(acc);
}

// fill motion_fea channels 126,127 with flow (bf16 + f32 copies)
__global__ void motion_tail_kernel(const float* __restrict__ flow,
                                   bf16_t* __restrict__ mb, float* __restrict__ mf)
{
  int pix = blockIdx.x * blockDim.x + threadIdx.x;
  if (pix >= NP) return;
  const int b = pix >> 12, sp = pix & (HW - 1);
#pragma unroll
  for (int c = 0; c < 2; ++c) {
    const float v = flow[(size_t)(b * 2 + c) * HW + sp];
    mb[(size_t)pix * 128 + 126 + c] = (bf16_t)v;
    mf[(size_t)pix * 128 + 126 + c] = v;
  }
}

// V [NP][128] -> VT [NB][128][HW]
__global__ void vt_kernel(const bf16_t* __restrict__ v, bf16_t* __restrict__ vT)
{
  int idx = blockIdx.x * blockDim.x + threadIdx.x;
  if (idx >= NP * 128) return;
  const int d = idx & 127;
  const int pix = idx >> 7;
  const int b = pix >> 12, j = pix & (HW - 1);
  vT[((size_t)(b * 128) + d) * HW + j] = v[(size_t)pix * 128 + d];
}

// conv1 input: [net | upfeat | motion | motion + gamma*agg]  -> [NP][512] bf16
__global__ void build_cat_kernel(const float* __restrict__ x1, const float* __restrict__ upfeat,
                                 const bf16_t* __restrict__ mb, const float* __restrict__ mf,
                                 const float* __restrict__ agg, const float* __restrict__ gamma,
                                 bf16_t* __restrict__ cat)
{
  int idx = blockIdx.x * blockDim.x + threadIdx.x;
  if (idx >= NP * 128) return;
  const int c = idx & 127;
  const int pix = idx >> 7;
  const int b = pix >> 12, sp = pix & (HW - 1);
  const size_t base = (size_t)pix * 512;
  cat[base + c]       = (bf16_t)x1[((size_t)(b * 256) + c) * HW + sp];            // net
  cat[base + 128 + c] = (bf16_t)upfeat[((size_t)(b * 128) + c) * HW + sp];        // upfeat
  cat[base + 256 + c] = mb[(size_t)pix * 128 + c];                                // motion
  cat[base + 384 + c] = (bf16_t)(mf[(size_t)pix * 128 + c] +
                                 gamma[0] * agg[(size_t)pix * 128 + c]);          // global
}

// ConvTranspose2d(32->16, k=4, stride=2, pad=1) + lrelu; X channels-last [NP][32]
__global__ void deconv_kernel(const float* __restrict__ X, const float* __restrict__ w,
                              const float* __restrict__ bias, float* __restrict__ up)
{
  int idx = blockIdx.x * blockDim.x + threadIdx.x;          // NB*16*128*128
  if (idx >= NB * 16 * 128 * 128) return;
  const int xo = idx & 127;
  const int yo = (idx >> 7) & 127;
  const int o  = (idx >> 14) & 15;
  const int b  = idx >> 18;
  float acc = bias[o];
  for (int ky = 0; ky < 4; ++ky) {
    const int yt = yo + ky - 2;
    if (yt < 0 || (yt & 1)) continue;
    const int yi = yt >> 1; if (yi >= WD) continue;
    for (int kx = 0; kx < 4; ++kx) {
      const int xt = xo + kx - 2;
      if (xt < 0 || (xt & 1)) continue;
      const int xi = xt >> 1; if (xi >= WD) continue;
      const float* xr = X + (size_t)(b * HW + yi * WD + xi) * 32;
      for (int i = 0; i < 32; ++i)
        acc += xr[i] * w[((size_t)(i * 16 + o) * 4 + (3 - ky)) * 4 + (3 - kx)];
    }
  }
  up[((size_t)(b * 16 + o) * 16384) + yo * 128 + xo] = lrelu_f(acc);
}

// trade: 3x3, 16 -> 64, 128x128, NCHW f32 in/out, no activation
__global__ void trade_kernel(const float* __restrict__ up, const float* __restrict__ w,
                             const float* __restrict__ bias, float* __restrict__ out)
{
  int idx = blockIdx.x * blockDim.x + threadIdx.x;          // NB*64*128*128
  if (idx >= NB * 64 * 128 * 128) return;
  const int xo = idx & 127;
  const int yo = (idx >> 7) & 127;
  const int o  = (idx >> 14) & 63;
  const int b  = idx >> 20;
  float acc = bias[o];
  for (int ky = 0; ky < 3; ++ky) {
    const int ys = yo + ky - 1; if (ys < 0 || ys >= 128) continue;
    for (int kx = 0; kx < 3; ++kx) {
      const int xs = xo + kx - 1; if (xs < 0 || xs >= 128) continue;
      for (int ic = 0; ic < 16; ++ic)
        acc += up[((size_t)(b * 16 + ic) * 16384) + ys * 128 + xs] *
               w[((o * 16 + ic) * 3 + ky) * 3 + kx];
    }
  }
  out[((size_t)(b * 64 + o) * 16384) + yo * 128 + xo] = acc;
}

// 3x3 conv, channels-last f32 input [NP][icStride], NCHW f32 output, no act
__global__ void head_conv_kernel(const float* __restrict__ X, int IC, int icStride,
                                 const float* __restrict__ w, const float* __restrict__ bias,
                                 float* __restrict__ out, int OC)
{
  int idx = blockIdx.x * blockDim.x + threadIdx.x;          // NP*OC
  if (idx >= NP * OC) return;
  const int oc  = idx % OC;
  const int pix = idx / OC;
  const int b = pix >> 12, sp = pix & (HW - 1);
  const int y = sp >> 6, x = sp & 63;
  float acc = bias[oc];
  for (int ky = 0; ky < 3; ++ky) {
    const int ys = y + ky - 1; if (ys < 0 || ys >= WD) continue;
    for (int kx = 0; kx < 3; ++kx) {
      const int xs = x + kx - 1; if (xs < 0 || xs >= WD) continue;
      const float* xr = X + (size_t)(b * HW + ys * WD + xs) * icStride;
      for (int ic = 0; ic < IC; ++ic)
        acc += xr[ic] * w[((oc * IC + ic) * 3 + ky) * 3 + kx];
    }
  }
  out[((size_t)(b * OC + oc) * HW) + sp] = acc;
}

// ---------------------------------------------------------------------------
// Launch
// ---------------------------------------------------------------------------
extern "C" void kernel_launch(void* const* d_in, const int* in_sizes, int n_in,
                              void* d_out, int out_size, void* d_ws, size_t ws_size,
                              hipStream_t stream)
{
  (void)in_sizes; (void)n_in; (void)out_size; (void)ws_size;
  const float* x1      = (const float*)d_in[0];
  const float* upfeat  = (const float*)d_in[1];
  const float* corr    = (const float*)d_in[2];
  const float* flow    = (const float*)d_in[3];
  const float* w_qk    = (const float*)d_in[4];
  const float* w_c1    = (const float*)d_in[5];   const float* b_c1 = (const float*)d_in[6];
  const float* w_c2    = (const float*)d_in[7];   const float* b_c2 = (const float*)d_in[8];
  const float* w_f1    = (const float*)d_in[9];   const float* b_f1 = (const float*)d_in[10];
  const float* w_f2    = (const float*)d_in[11];  const float* b_f2 = (const float*)d_in[12];
  const float* w_mo    = (const float*)d_in[13];  const float* b_mo = (const float*)d_in[14];
  const float* w_v     = (const float*)d_in[15];
  const float* gamma   = (const float*)d_in[16];
  const float* w1      = (const float*)d_in[17];  const float* b1 = (const float*)d_in[18];
  const float* w2      = (const float*)d_in[19];  const float* b2 = (const float*)d_in[20];
  const float* w3      = (const float*)d_in[21];  const float* b3 = (const float*)d_in[22];
  const float* w4      = (const float*)d_in[23];  const float* b4 = (const float*)d_in[24];
  const float* w_up    = (const float*)d_in[25];  const float* b_up = (const float*)d_in[26];
  const float* w_tr    = (const float*)d_in[27];  const float* b_tr = (const float*)d_in[28];
  const float* w_fl    = (const float*)d_in[29];  const float* b_fl = (const float*)d_in[30];
  const float* w_mk    = (const float*)d_in[31];  const float* b_mk = (const float*)d_in[32];

  float* out = (float*)d_out;
  float* out_trade = out;                                   // (4,64,128,128)
  float* out_up    = out + (size_t)4 * 64 * 16384;          // (4,16,128,128)
  float* out_flow  = out_up + (size_t)4 * 16 * 16384;       // (4,2,64,64)
  float* out_mask  = out_flow + (size_t)4 * 2 * HW;         // (4,1,64,64)

  // workspace carve-out
  char* wsb = (char*)d_ws;
  size_t off = 0;
  auto alloc = [&](size_t bytes) -> void* {
    void* p = wsb + off;
    off = (off + bytes + 255) & ~(size_t)255;
    return p;
  };
  bf16_t* zpg   = (bf16_t*)alloc((size_t)2048 * 2);   // zero page for OOB taps
  bf16_t* wp_qk = (bf16_t*)alloc((size_t)256 * 128 * 2);
  bf16_t* wp_c1 = (bf16_t*)alloc((size_t)256 * 96 * 2);
  bf16_t* wp_c2 = (bf16_t*)alloc((size_t)192 * 9 * 256 * 2);
  bf16_t* wp_f2 = (bf16_t*)alloc((size_t)64 * 9 * 128 * 2);
  bf16_t* wp_mo = (bf16_t*)alloc((size_t)128 * 9 * 256 * 2);
  bf16_t* wp_v  = (bf16_t*)alloc((size_t)128 * 128 * 2);
  bf16_t* wp_1  = (bf16_t*)alloc((size_t)128 * 9 * 512 * 2);
  bf16_t* wp_2  = (bf16_t*)alloc((size_t)96 * 9 * 128 * 2);
  bf16_t* wp_3  = (bf16_t*)alloc((size_t)64 * 9 * 96 * 2);
  bf16_t* wp_4  = (bf16_t*)alloc((size_t)32 * 9 * 64 * 2);
  bf16_t* inp_bf  = (bf16_t*)alloc((size_t)NP * 128 * 2);
  bf16_t* qk_out  = (bf16_t*)alloc((size_t)NP * 256 * 2);
  bf16_t* corr_bf = (bf16_t*)alloc((size_t)NP * 96 * 2);
  bf16_t* cor1    = (bf16_t*)alloc((size_t)NP * 256 * 2);
  bf16_t* corflo  = (bf16_t*)alloc((size_t)NP * 256 * 2);
  bf16_t* flo1    = (bf16_t*)alloc((size_t)NP * 128 * 2);
  bf16_t* mot_bf  = (bf16_t*)alloc((size_t)NP * 128 * 2);
  float*  mot_f   = (float* )alloc((size_t)NP * 128 * 4);
  bf16_t* vbuf    = (bf16_t*)alloc((size_t)NP * 128 * 2);
  bf16_t* vT      = (bf16_t*)alloc((size_t)NP * 128 * 2);
  float*  agg     = (float* )alloc((size_t)NP * 128 * 4);
  bf16_t* xcat    = (bf16_t*)alloc((size_t)NP * 512 * 2);
  bf16_t* x1b     = (bf16_t*)alloc((size_t)NP * 128 * 2);
  bf16_t* x2b     = (bf16_t*)alloc((size_t)NP * 96 * 2);
  bf16_t* x3b     = (bf16_t*)alloc((size_t)NP * 64 * 2);
  bf16_t* x4b     = (bf16_t*)alloc((size_t)NP * 32 * 2);
  float*  x4f     = (float* )alloc((size_t)NP * 32 * 4);

  const int T = 256;
  auto g1 = [&](int n) { return dim3((n + T - 1) / T); };

  // zero page
  zero_fill_kernel<<<g1(2048), T, 0, stream>>>(zpg, 2048);

  // ---- weight packing (OIHW f32 -> [OCpad][k][k][ICpad] bf16) ----
  { int n = 256*1*128;   pack_w_kernel<<<g1(n), T, 0, stream>>>(w_qk, wp_qk, 256, 256, 128, 128, 1, n); }
  { int n = 256*1*96;    pack_w_kernel<<<g1(n), T, 0, stream>>>(w_c1, wp_c1, 256, 256, 81, 96, 1, n); }
  { int n = 192*9*256;   pack_w_kernel<<<g1(n), T, 0, stream>>>(w_c2, wp_c2, 192, 192, 256, 256, 9, n); }
  { int n = 64*9*128;    pack_w_kernel<<<g1(n), T, 0, stream>>>(w_f2, wp_f2, 64, 64, 128, 128, 9, n); }
  { int n = 128*9*256;   pack_w_kernel<<<g1(n), T, 0, stream>>>(w_mo, wp_mo, 126, 128, 256, 256, 9, n); }
  { int n = 128*1*128;   pack_w_kernel<<<g1(n), T, 0, stream>>>(w_v,  wp_v,  128, 128, 128, 128, 1, n); }
  { int n = 128*9*512;   pack_w_kernel<<<g1(n), T, 0, stream>>>(w1,   wp_1,  128, 128, 512, 512, 9, n); }
  { int n = 96*9*128;    pack_w_kernel<<<g1(n), T, 0, stream>>>(w2,   wp_2,  96, 96, 128, 128, 9, n); }
  { int n = 64*9*96;     pack_w_kernel<<<g1(n), T, 0, stream>>>(w3,   wp_3,  64, 64, 96, 96, 9, n); }
  { int n = 32*9*64;     pack_w_kernel<<<g1(n), T, 0, stream>>>(w4,   wp_4,  32, 32, 64, 64, 9, n); }

  // ---- activation packing ----
  { int n = NP*128; pack_x_kernel<<<g1(n), T, 0, stream>>>(x1,   inp_bf,  256, 128, 128, 128, n); }
  { int n = NP*96;  pack_x_kernel<<<g1(n), T, 0, stream>>>(corr, corr_bf, 81, 0, 81, 96, n); }

  const dim3 wblk(32);
  const int PT = NP / 16;   // 1024 pixel tiles

  // qk = 1x1 conv (no bias/act) -> [NP][256]    (OCT=4 -> grid.y = 4)
  wmma_conv_kernel<4><<<dim3(PT, 4), wblk, 0, stream>>>(inp_bf, 128, zpg, wp_qk, nullptr,
      qk_out, nullptr, 256, 0, 256, 1, 1, 1, 0);
  // cor = lrelu(convc1(corr)) : 1x1, 81->256
  wmma_conv_kernel<4><<<dim3(PT, 4), wblk, 0, stream>>>(corr_bf, 96, zpg, wp_c1, b_c1,
      cor1, nullptr, 256, 0, 256, 1, 1, 1, 1);
  // cor = lrelu(convc2(cor)) : 3x3, 256->192  -> corflo ch 0..191
  wmma_conv_kernel<3><<<dim3(PT, 4), wblk, 0, stream>>>(cor1, 256, zpg, wp_c2, b_c2,
      corflo, nullptr, 256, 0, 192, 3, 3, 1, 1);
  // flo = lrelu(convf1(flow)) : 7x7, 2->128 (plain)
  convf1_kernel<<<g1(NP*128), T, 0, stream>>>(flow, w_f1, b_f1, flo1);
  // flo = lrelu(convf2(flo)) : 3x3, 128->64 -> corflo ch 192..255 (free concat)
  wmma_conv_kernel<4><<<dim3(PT, 1), wblk, 0, stream>>>(flo1, 128, zpg, wp_f2, b_f2,
      corflo, nullptr, 256, 192, 64, 3, 3, 1, 1);
  // motion = lrelu(conv_motion(cor_flo)) : 3x3, 256->126, + f32 copy
  wmma_conv_kernel<4><<<dim3(PT, 2), wblk, 0, stream>>>(corflo, 256, zpg, wp_mo, b_mo,
      mot_bf, mot_f, 128, 0, 126, 3, 3, 1, 1);
  // motion ch 126,127 = flow
  motion_tail_kernel<<<g1(NP), T, 0, stream>>>(flow, mot_bf, mot_f);
  // v = agg_to_v(motion) : 1x1, 128->128, no bias/act
  wmma_conv_kernel<4><<<dim3(PT, 2), wblk, 0, stream>>>(mot_bf, 128, zpg, wp_v, nullptr,
      vbuf, nullptr, 128, 0, 128, 1, 1, 1, 0);
  // transpose V for contiguous B-fragment loads
  vt_kernel<<<g1(NP*128), T, 0, stream>>>(vbuf, vT);
  // agg = softmax(scale*Q^T K) V   (flash, never materializes 4096x4096)
  flash_attn_kernel<<<dim3(HW/16, NB), wblk, 0, stream>>>(qk_out, vT, agg);
  // conv1 input concat
  build_cat_kernel<<<g1(NP*128), T, 0, stream>>>(x1, upfeat, mot_bf, mot_f, agg, gamma, xcat);

  // conv stack
  wmma_conv_kernel<4><<<dim3(PT, 2), wblk, 0, stream>>>(xcat, 512, zpg, wp_1, b1,
      x1b, nullptr, 128, 0, 128, 3, 3, 1, 1);
  wmma_conv_kernel<3><<<dim3(PT, 2), wblk, 0, stream>>>(x1b, 128, zpg, wp_2, b2,
      x2b, nullptr, 96, 0, 96, 3, 3, 4, 1);
  wmma_conv_kernel<4><<<dim3(PT, 1), wblk, 0, stream>>>(x2b, 96, zpg, wp_3, b3,
      x3b, nullptr, 64, 0, 64, 3, 3, 8, 1);
  wmma_conv_kernel<2><<<dim3(PT, 1), wblk, 0, stream>>>(x3b, 64, zpg, wp_4, b4,
      x4b, x4f, 32, 0, 32, 3, 3, 1, 1);

  // heads
  deconv_kernel<<<g1(NB*16*128*128), T, 0, stream>>>(x4f, w_up, b_up, out_up);
  trade_kernel<<<g1(NB*64*128*128), T, 0, stream>>>(out_up, w_tr, b_tr, out_trade);
  head_conv_kernel<<<g1(NP*2), T, 0, stream>>>(x4f, 32, 32, w_fl, b_fl, out_flow, 2);
  head_conv_kernel<<<g1(NP*1), T, 0, stream>>>(x4f, 32, 32, w_mk, b_mk, out_mask, 1);
}
